// Gumbel_Token_Selection_Block_Wrapper_44152263803182
// MI455X (gfx1250) — matrix operand order repaired and checked by reading, not compiled, not run on hardware
//
#include <hip/hip_runtime.h>
#include <hip/hip_bf16.h>
#include <stdint.h>

// ---------------- problem constants ----------------
constexpr int BATCH   = 64;
constexpr int SEQ     = 577;
constexpr int DIM     = 768;
constexpr int HEADS   = 12;
constexpr int DHEAD   = 64;
constexpr int DFFN    = 3072;
constexpr int MROWS   = BATCH * SEQ;      // 36928 = 2308*16
constexpr int MTILES  = MROWS / 16;       // 2308 = 4*577
constexpr int MBLOCKS = MTILES / 4;       // 577 (64-row blocks)
constexpr int NPATCH  = 576;
constexpr int NALPHA  = 288;
constexpr int MCS     = 16;
constexpr float MCTAU_INV = 2.0f;         // 1/0.5
constexpr float EPSV  = 1e-8f;

typedef _Float16 half_t;
typedef __attribute__((ext_vector_type(16))) _Float16 v16h;
typedef __attribute__((ext_vector_type(8)))  _Float16 v8h;
typedef __attribute__((ext_vector_type(8)))  float    v8f;
typedef unsigned int v4u __attribute__((ext_vector_type(4)));
typedef int          v4i __attribute__((ext_vector_type(4)));
typedef int          v8i __attribute__((ext_vector_type(8)));

// ---------------- WMMA fragment helpers (CDNA5 wave32 layouts) ----------------
// A (16x32 f16): lane L -> row L&15; halves e=0..7 -> K=koff+e, e=8..15 -> K=16+koff+(e-8),
// koff = 8*(L>=16).  rowptr points at K=0 of the 32-wide chunk for this lane's row.
static __device__ inline v16h a_frag(const half_t* rowptr, int lane) {
    int koff = (lane >> 4) << 3;
    v8h lo = *(const v8h*)(rowptr + koff);
    v8h hi = *(const v8h*)(rowptr + 16 + koff);
    return __builtin_shufflevector(lo, hi, 0,1,2,3,4,5,6,7,8,9,10,11,12,13,14,15);
}
// B (32x16 f16): lane L -> col L&15; halves e=0..7 -> K=koff+e, e=8..15 -> K=16+koff+(e-8), koff = 8*(L>=16).
// colptr points at K=0 of this lane's column (column-contiguous storage).
static __device__ inline v16h b_frag(const half_t* colptr, int lane) {
    int koff = (lane >> 4) << 3;
    v8h lo = *(const v8h*)(colptr + koff);
    v8h hi = *(const v8h*)(colptr + 16 + koff);
    return __builtin_shufflevector(lo, hi, 0,1,2,3,4,5,6,7,8,9,10,11,12,13,14,15);
}
static __device__ inline v8f wmma32(v16h a, v16h b, v8f c) {
    return __builtin_amdgcn_wmma_f32_16x16x32_f16(false, a, false, b, (short)0, c, false, false);
}

static __device__ inline float gelu_tanh(float v) {
    float t = 0.7978845608f * (v + 0.044715f * v * v * v);
    return 0.5f * v * (1.0f + tanhf(t));
}

// ---------------- Tensor Data Mover: 2D tile (tile_d0 x tile_d1) global->LDS ----------------
// Wt layout [rows][K] f16; loads tile_d1 rows of tile_d0 contiguous elements, row stride = dim0.
static __device__ inline unsigned lds_off(const void* p) {
    return (unsigned)(unsigned long)p;   // flat LDS addr: low 32 bits = LDS byte offset
}
static __device__ inline void tdm_load_2d(const void* gaddr, unsigned lds_addr,
                                          unsigned tile_d0, unsigned tile_d1,
                                          unsigned dim0_elems, unsigned dim1_elems) {
    unsigned long ga = (unsigned long)gaddr;
    v4u g0;
    g0.x = 1u;                                        // count=1 (valid descriptor)
    g0.y = lds_addr;                                  // lds_addr [63:32]
    g0.z = (unsigned)(ga & 0xFFFFFFFFu);              // global_addr[31:0]
    g0.w = (unsigned)((ga >> 32) & 0x01FFFFFFu) | 0x80000000u;  // addr[56:32] | type=2
    v8i g1;
    g1[0] = (int)0x10000u;                            // data_size=1 (2 bytes)
    g1[1] = (int)((dim0_elems & 0xFFFFu) << 16);      // tensor_dim0[15:0]
    g1[2] = (int)(((dim0_elems >> 16) & 0xFFFFu) | ((dim1_elems & 0xFFFFu) << 16));
    g1[3] = (int)(((dim1_elems >> 16) & 0xFFFFu) | (tile_d0 << 16));   // tile_dim0
    g1[4] = (int)tile_d1;                             // tile_dim1 (tile_dim2=0)
    g1[5] = (int)dim0_elems;                          // tensor_dim0_stride[31:0] = K
    g1[6] = 0;                                        // stride0 hi / stride1 lo
    g1[7] = 0;
    v4i z4 = {0, 0, 0, 0};
#if defined(__clang_major__) && (__clang_major__ >= 23)
    v8i z8 = {0, 0, 0, 0, 0, 0, 0, 0};
    __builtin_amdgcn_tensor_load_to_lds(g0, g1, z4, z4, z8, 0);
#else
    __builtin_amdgcn_tensor_load_to_lds(g0, g1, z4, z4, 0);
#endif
}

// ---------------- weight convert + transpose: w[K][N] f32 -> wt[N][K] f16 ----------------
__global__ void transpose_cast_kernel(const float* __restrict__ w, half_t* __restrict__ wt,
                                      int K, int Ncols) {
    long idx = (long)blockIdx.x * 256 + threadIdx.x;
    long total = (long)K * Ncols;
    if (idx >= total) return;
    int n = (int)(idx % Ncols);
    long k = idx / Ncols;
    wt[(long)n * K + k] = (half_t)w[idx];
}

// ---------------- LayerNorm + cast to f16 ----------------
__global__ __launch_bounds__(256)
void layernorm_cast_kernel(const float* __restrict__ x, const float* __restrict__ g,
                           const float* __restrict__ bta, half_t* __restrict__ out) {
    __shared__ float red[256];
    long row = blockIdx.x;
    const float* xr = x + row * DIM;
    int tid = threadIdx.x;
    float s = 0.f;
    for (int i = tid; i < DIM; i += 256) s += xr[i];
    red[tid] = s; __syncthreads();
    for (int o = 128; o > 0; o >>= 1) { if (tid < o) red[tid] += red[tid + o]; __syncthreads(); }
    float mu = red[0] * (1.0f / DIM);
    __syncthreads();
    float v = 0.f;
    for (int i = tid; i < DIM; i += 256) { float d = xr[i] - mu; v += d * d; }
    red[tid] = v; __syncthreads();
    for (int o = 128; o > 0; o >>= 1) { if (tid < o) red[tid] += red[tid + o]; __syncthreads(); }
    float rstd = rsqrtf(red[0] * (1.0f / DIM) + 1e-6f);
    for (int i = tid; i < DIM; i += 256)
        out[row * DIM + i] = (half_t)((xr[i] - mu) * rstd * g[i] + bta[i]);
}

// ---------------- WMMA GEMM with TDM-staged weights ----------------
// C = A[M x K] * Wt^T, Wt stored [Ncols][K]. Block = 64M x 128N (8 waves: 4 m-tiles x 2 n-tiles).
// B tile (128 cols x 32 k) is DMA'd into LDS by the Tensor Data Mover (wave 0), double buffered,
// overlapped with the 4 WMMAs/wave per k-step. Weight L2 traffic drops 4x vs per-wave loads.
// EPI: 0 = bias -> f16 out; 1 = resid + ls*(acc+bias) -> f32 out; 2 = gelu(acc+bias) -> f16 out
template<int EPI>
__global__ __launch_bounds__(256)
void gemm_wmma_kernel(const half_t* __restrict__ A, const half_t* __restrict__ Wt,
                      const float* __restrict__ bias, const float* __restrict__ ls,
                      const float* __restrict__ resid,
                      half_t* __restrict__ out16, float* __restrict__ out32,
                      int K, int Ncols, int numNb) {
    __shared__ half_t Bt[2][128 * 32];    // [buf][n_local * 32 + k_local], 8KB each
    int wid  = threadIdx.x >> 5;
    int lane = threadIdx.x & 31;
    int mb = blockIdx.x / numNb;
    int nb = blockIdx.x % numNb;
    if (mb >= MBLOCKS) return;
    int mi = wid & 3, ni = wid >> 2;
    int m0 = mb * 64 + mi * 16;           // wave rows
    int n0 = nb * 128 + ni * 64;          // wave cols (global)
    int nl = ni * 64;                     // wave cols (local in B tile)
    const half_t* arow   = A  + (long)(m0 + (lane & 15)) * K;
    const char*   wtile  = (const char*)(Wt + (long)nb * 128 * K);  // tile row 0, k 0
    unsigned lds0 = lds_off(&Bt[0][0]);
    unsigned lds1 = lds_off(&Bt[1][0]);
    int ksteps = K >> 5;

    if (wid == 0) {
        tdm_load_2d(wtile, lds0, 32, 128, (unsigned)K, 128u);
        __builtin_amdgcn_s_wait_tensorcnt(0);
    }
    __syncthreads();

    v8f acc[4] = {};
    for (int s = 0; s < ksteps; ++s) {
        if (wid == 0 && s + 1 < ksteps)
            tdm_load_2d(wtile + (long)(s + 1) * 64 /* 32 elems * 2B */,
                        (s & 1) ? lds0 : lds1, 32, 128, (unsigned)K, 128u);
        const half_t* bt = &Bt[s & 1][0];
        v16h a = a_frag(arow + s * 32, lane);
        #pragma unroll
        for (int t = 0; t < 4; ++t) {
            v16h b = b_frag(bt + (nl + t * 16 + (lane & 15)) * 32, lane);
            acc[t] = wmma32(a, b, acc[t]);
        }
        if (wid == 0 && s + 1 < ksteps)
            __builtin_amdgcn_s_wait_tensorcnt(0);
        __syncthreads();
    }

    int rbase = (lane >> 4) * 8;
    #pragma unroll
    for (int t = 0; t < 4; ++t) {
        #pragma unroll
        for (int i = 0; i < 8; ++i) {
            int r = m0 + rbase + i;
            int c = n0 + t * 16 + (lane & 15);
            float v = acc[t][i] + bias[c];
            long o = (long)r * Ncols + c;
            if constexpr (EPI == 0)      out16[o] = (half_t)v;
            else if constexpr (EPI == 1) out32[o] = resid[o] + ls[c] * v;
            else                         out16[o] = (half_t)gelu_tanh(v);
        }
    }
}

// ---------------- attention: one wave per (b, h, 16-query tile) ----------------
// qkv16: [BATCH*SEQ][3*DIM] f16, q at col 0, k at +DIM, v at +2*DIM (per head h*64+dh)
__global__ __launch_bounds__(32)
void attention_kernel(const half_t* __restrict__ qkv, half_t* __restrict__ o16,
                      float* __restrict__ cls_rows) {
    __shared__ float  S[16][608];     // scores strip, padded (37*16=592 -> 608)
    __shared__ half_t Vt[64][32];     // transposed V tile: [dh][key]
    int lane = threadIdx.x;
    int qt = blockIdx.x, h = blockIdx.y, b = blockIdx.z;
    int q0 = qt * 16;
    int mrow = q0 + (lane & 15); if (mrow > SEQ - 1) mrow = SEQ - 1;
    const half_t* qrow = qkv + (long)(b * SEQ + mrow) * (3 * DIM) + h * DHEAD;

    // --- Phase 1: S = (Q K^T) / 8 via WMMA (K-dim = 64 -> 2 chained wmmas) ---
    for (int kt = 0; kt < 37; ++kt) {
        int kn = kt * 16 + (lane & 15); if (kn > SEQ - 1) kn = SEQ - 1;
        const half_t* krow = qkv + (long)(b * SEQ + kn) * (3 * DIM) + DIM + h * DHEAD;
        v8f acc = {};
        acc = wmma32(a_frag(qrow, lane),      b_frag(krow, lane),      acc);
        acc = wmma32(a_frag(qrow + 32, lane), b_frag(krow + 32, lane), acc);
        int col = kt * 16 + (lane & 15);
        int rbase = (lane >> 4) * 8;
        #pragma unroll
        for (int i = 0; i < 8; ++i) S[rbase + i][col] = acc[i] * 0.125f;
    }
    __syncthreads();
    // mask padded key columns (577..607)
    for (int t = lane; t < 16 * 31; t += 32) {
        int r = t / 31, c = SEQ + t % 31;
        S[r][c] = -1e30f;
    }
    __syncthreads();

    // --- Phase 2: softmax per row; lanes (L, L+16) cooperate on row L&15 ---
    int row = lane & 15, start = lane >> 4;
    float mx = -1e30f;
    for (int c = start; c < 608; c += 2) mx = fmaxf(mx, S[row][c]);
    mx = fmaxf(mx, __shfl_xor(mx, 16, 32));
    float sum = 0.f;
    for (int c = start; c < 608; c += 2) { float p = __expf(S[row][c] - mx); S[row][c] = p; sum += p; }
    sum += __shfl_xor(sum, 16, 32);
    float inv = 1.0f / sum;
    for (int c = start; c < 608; c += 2) S[row][c] *= inv;
    __syncthreads();
    // CLS attention row (query 0) -> per-head buffer (deterministic reduction later)
    if (qt == 0 && row == 0) {
        float* dst = cls_rows + ((long)b * HEADS + h) * SEQ;
        for (int c = start; c < SEQ; c += 2) dst[c] = S[0][c];
    }

    // --- Phase 3: O = P V via WMMA (19 key-steps of 32, V staged transposed in LDS) ---
    v8f oacc[4] = {};
    for (int ks = 0; ks < 19; ++ks) {
        for (int t = lane; t < 64 * 32; t += 32) {
            int dh = t >> 5, kl = t & 31;
            int tok = ks * 32 + kl;
            half_t val = (half_t)0.f;
            if (tok < SEQ)
                val = qkv[(long)(b * SEQ + tok) * (3 * DIM) + 2 * DIM + h * DHEAD + dh];
            Vt[dh][kl] = val;
        }
        __syncthreads();
        int koff = (lane >> 4) << 3;
        v16h pa;
        #pragma unroll
        for (int e = 0; e < 8; ++e) pa[e]     = (half_t)S[row][ks * 32 + koff + e];
        #pragma unroll
        for (int e = 0; e < 8; ++e) pa[8 + e] = (half_t)S[row][ks * 32 + 16 + koff + e];
        #pragma unroll
        for (int t = 0; t < 4; ++t) {
            v16h bf = b_frag(&Vt[t * 16 + (lane & 15)][0], lane);
            oacc[t] = wmma32(pa, bf, oacc[t]);
        }
        __syncthreads();
    }
    int rbase = (lane >> 4) * 8;
    #pragma unroll
    for (int t = 0; t < 4; ++t)
        #pragma unroll
        for (int i = 0; i < 8; ++i) {
            int qg = q0 + rbase + i;
            if (qg < SEQ) {
                int dh = t * 16 + (lane & 15);
                o16[(long)(b * SEQ + qg) * DIM + h * DHEAD + dh] = (half_t)oacc[t][i];
            }
        }
}

// ---------------- CLS attention head-mean ----------------
__global__ void cls_reduce_kernel(const float* __restrict__ cls_rows, float* __restrict__ cls) {
    int idx = blockIdx.x * 256 + threadIdx.x;
    if (idx >= BATCH * SEQ) return;
    int b = idx / SEQ, c = idx % SEQ;
    float s = 0.f;
    for (int h = 0; h < HEADS; ++h) s += cls_rows[((long)b * HEADS + h) * SEQ + c];
    cls[idx] = s * (1.0f / HEADS);
}

// ---------------- Gumbel MC scores ----------------
static __device__ inline float rng_uniform(uint32_t a, uint32_t b, uint32_t c) {
    uint32_t h = a * 0x9E3779B1u ^ b * 0x85EBCA77u ^ c * 0xC2B2AE3Du ^ 0xB5297A4Du;
    h ^= h >> 16; h *= 0x7FEB352Du; h ^= h >> 15; h *= 0x846CA68Bu; h ^= h >> 16;
    float u = (float)(h >> 8) * (1.0f / 16777216.0f);
    return fminf(fmaxf(u, 1e-6f), 1.0f - 1e-6f);
}

__global__ __launch_bounds__(256)
void scores_kernel(const float* __restrict__ cls, float* __restrict__ scores) {
    __shared__ float red[256];
    int b = blockIdx.x, tid = threadIdx.x;
    float lg[3]; int idx[3]; int cnt = 0;
    for (int i = tid; i < NPATCH; i += 256) {
        lg[cnt] = logf(cls[b * SEQ + 1 + i] + EPSV);
        idx[cnt] = i; ++cnt;
    }
    float acc[3] = {0.f, 0.f, 0.f};
    for (int s = 0; s < MCS; ++s) {
        float z[3]; float lmx = -1e30f;
        for (int j = 0; j < cnt; ++j) {
            float u = rng_uniform((uint32_t)s, (uint32_t)b, (uint32_t)idx[j]);
            float gm = -logf(-logf(u));
            z[j] = (lg[j] + gm) * MCTAU_INV;
            lmx = fmaxf(lmx, z[j]);
        }
        red[tid] = lmx; __syncthreads();
        for (int o = 128; o > 0; o >>= 1) { if (tid < o) red[tid] = fmaxf(red[tid], red[tid + o]); __syncthreads(); }
        float mx = red[0]; __syncthreads();
        float ps = 0.f;
        for (int j = 0; j < cnt; ++j) { z[j] = __expf(z[j] - mx); ps += z[j]; }
        red[tid] = ps; __syncthreads();
        for (int o = 128; o > 0; o >>= 1) { if (tid < o) red[tid] += red[tid + o]; __syncthreads(); }
        float inv = 1.0f / red[0]; __syncthreads();
        for (int j = 0; j < cnt; ++j) acc[j] += z[j] * inv * (1.0f / MCS);
    }
    for (int j = 0; j < cnt; ++j) scores[b * NPATCH + idx[j]] = acc[j];
}

// ---------------- bitonic top-k (288 of 576) + ascending index sort ----------------
__global__ __launch_bounds__(512)
void topk_sort_kernel(const float* __restrict__ scores, int* __restrict__ rel_sorted) {
    __shared__ float key[1024];
    __shared__ int   id[1024];
    __shared__ int   sidx[512];
    int b = blockIdx.x, tid = threadIdx.x;
    for (int i = tid; i < 1024; i += 512) {
        key[i] = (i < NPATCH) ? scores[b * NPATCH + i] : -1e30f;
        id[i]  = i;
    }
    __syncthreads();
    for (int k = 2; k <= 1024; k <<= 1)
        for (int j = k >> 1; j > 0; j >>= 1) {
            for (int i = tid; i < 1024; i += 512) {
                int ixj = i ^ j;
                if (ixj > i) {
                    bool desc = ((i & k) == 0);
                    bool sw = desc ? (key[i] < key[ixj]) : (key[i] > key[ixj]);
                    if (sw) {
                        float tk = key[i]; key[i] = key[ixj]; key[ixj] = tk;
                        int   ti = id[i];  id[i]  = id[ixj];  id[ixj]  = ti;
                    }
                }
            }
            __syncthreads();
        }
    sidx[tid] = (tid < NALPHA) ? id[tid] : 0x7fffffff;
    __syncthreads();
    for (int k = 2; k <= 512; k <<= 1)
        for (int j = k >> 1; j > 0; j >>= 1) {
            int i = tid, ixj = i ^ j;
            if (ixj > i) {
                bool asc = ((i & k) == 0);
                bool sw = asc ? (sidx[i] > sidx[ixj]) : (sidx[i] < sidx[ixj]);
                if (sw) { int t = sidx[i]; sidx[i] = sidx[ixj]; sidx[ixj] = t; }
            }
            __syncthreads();
        }
    if (tid < NALPHA) rel_sorted[b * NALPHA + tid] = sidx[tid];
}

// ---------------- gather selected tokens + scores into output ----------------
__global__ __launch_bounds__(256)
void gather_kernel(const float* __restrict__ x2, const float* __restrict__ scores,
                   const int* __restrict__ rel_sorted, float* __restrict__ out) {
    int bj = blockIdx.x;
    int b = bj / (NALPHA + 1), j = bj % (NALPHA + 1);
    int rel = (j == 0) ? 0 : rel_sorted[b * NALPHA + j - 1];
    int tok = (j == 0) ? 0 : rel + 1;
    const float* src = x2 + (long)(b * SEQ + tok) * DIM;
    float* dst = out + (long)bj * DIM;
    for (int c = threadIdx.x; c < DIM; c += 256) dst[c] = src[c];
    if (threadIdx.x == 0)
        out[(long)BATCH * (NALPHA + 1) * DIM + bj] =
            (j == 0) ? 1.0f : scores[b * NPATCH + rel];
}

// ---------------- host ----------------
extern "C" void kernel_launch(void* const* d_in, const int* in_sizes, int n_in,
                              void* d_out, int out_size, void* d_ws, size_t ws_size,
                              hipStream_t stream) {
    (void)in_sizes; (void)n_in; (void)out_size; (void)ws_size;
    const float* x      = (const float*)d_in[0];
    const float* g1     = (const float*)d_in[1];
    const float* b1     = (const float*)d_in[2];
    const float* w_qkv  = (const float*)d_in[3];
    const float* b_qkv  = (const float*)d_in[4];
    const float* w_proj = (const float*)d_in[5];
    const float* b_proj = (const float*)d_in[6];
    const float* ls1    = (const float*)d_in[7];
    const float* g2     = (const float*)d_in[8];
    const float* b2     = (const float*)d_in[9];
    const float* w_fc1  = (const float*)d_in[10];
    const float* b_fc1  = (const float*)d_in[11];
    const float* w_fc2  = (const float*)d_in[12];
    const float* b_fc2  = (const float*)d_in[13];
    const float* ls2    = (const float*)d_in[14];
    float* out = (float*)d_out;

    char* ws = (char*)d_ws;
    size_t off = 0;
    auto alloc = [&](size_t bytes) -> char* {
        off = (off + 255) & ~(size_t)255;
        char* p = ws + off; off += bytes; return p;
    };
    half_t* h16    = (half_t*)alloc((size_t)MROWS * DIM * 2);
    half_t* wtq    = (half_t*)alloc((size_t)(3 * DIM) * DIM * 2);
    half_t* wtp    = (half_t*)alloc((size_t)DIM * DIM * 2);
    half_t* wtf1   = (half_t*)alloc((size_t)DFFN * DIM * 2);
    half_t* wtf2   = (half_t*)alloc((size_t)DIM * DFFN * 2);
    half_t* r1     = (half_t*)alloc((size_t)MROWS * DFFN * 2); // qkv16 (dies) then fc1_16
    half_t* qkv16  = r1;
    half_t* fc116  = r1;
    half_t* o16    = (half_t*)alloc((size_t)MROWS * DIM * 2);
    float*  x1     = (float*)alloc((size_t)MROWS * DIM * 4);
    float*  x2     = (float*)alloc((size_t)MROWS * DIM * 4);
    float*  clsr   = (float*)alloc((size_t)BATCH * HEADS * SEQ * 4);
    float*  clsa   = (float*)alloc((size_t)BATCH * SEQ * 4);
    float*  scores = (float*)alloc((size_t)BATCH * NPATCH * 4);
    int*    rel    = (int*)alloc((size_t)BATCH * NALPHA * 4);

    // 1) convert+transpose weights to f16 [N][K]
    {
        long t;
        t = (long)DIM * (3 * DIM);
        transpose_cast_kernel<<<(t + 255) / 256, 256, 0, stream>>>(w_qkv, wtq, DIM, 3 * DIM);
        t = (long)DIM * DIM;
        transpose_cast_kernel<<<(t + 255) / 256, 256, 0, stream>>>(w_proj, wtp, DIM, DIM);
        t = (long)DIM * DFFN;
        transpose_cast_kernel<<<(t + 255) / 256, 256, 0, stream>>>(w_fc1, wtf1, DIM, DFFN);
        t = (long)DFFN * DIM;
        transpose_cast_kernel<<<(t + 255) / 256, 256, 0, stream>>>(w_fc2, wtf2, DFFN, DIM);
    }
    // 2) LN1
    layernorm_cast_kernel<<<MROWS, 256, 0, stream>>>(x, g1, b1, h16);
    // 3) QKV GEMM (f16 out)
    {
        int numNb = (3 * DIM) / 128;
        gemm_wmma_kernel<0><<<MBLOCKS * numNb, 256, 0, stream>>>(
            h16, wtq, b_qkv, nullptr, nullptr, qkv16, nullptr, DIM, 3 * DIM, numNb);
    }
    // 4) attention + CLS reduce
    attention_kernel<<<dim3(37, HEADS, BATCH), 32, 0, stream>>>(qkv16, o16, clsr);
    cls_reduce_kernel<<<(BATCH * SEQ + 255) / 256, 256, 0, stream>>>(clsr, clsa);
    // 5) proj GEMM + layerscale residual -> x1
    {
        int numNb = DIM / 128;
        gemm_wmma_kernel<1><<<MBLOCKS * numNb, 256, 0, stream>>>(
            o16, wtp, b_proj, ls1, x, nullptr, x1, DIM, DIM, numNb);
    }
    // 6) LN2
    layernorm_cast_kernel<<<MROWS, 256, 0, stream>>>(x1, g2, b2, h16);
    // 7) FC1 GEMM + GELU (f16 out, reuses qkv region)
    {
        int numNb = DFFN / 128;
        gemm_wmma_kernel<2><<<MBLOCKS * numNb, 256, 0, stream>>>(
            h16, wtf1, b_fc1, nullptr, nullptr, fc116, nullptr, DIM, DFFN, numNb);
    }
    // 8) FC2 GEMM + layerscale residual -> x2
    {
        int numNb = DIM / 128;
        gemm_wmma_kernel<1><<<MBLOCKS * numNb, 256, 0, stream>>>(
            fc116, wtf2, b_fc2, ls2, x1, nullptr, x2, DFFN, DIM, numNb);
    }
    // 9) Gumbel MC scores, top-k, gather
    scores_kernel<<<BATCH, 256, 0, stream>>>(clsa, scores);
    topk_sort_kernel<<<BATCH, 512, 0, stream>>>(scores, rel);
    gather_kernel<<<BATCH * (NALPHA + 1), 256, 0, stream>>>(x2, scores, rel, out);
}
